// MultiHeadAttention_2448131359429
// MI455X (gfx1250) — matrix operand order repaired and checked by reading, not compiled, not run on hardware
//
#include <hip/hip_runtime.h>

// ---------------------------------------------------------------------------
// MHA for MI455X (gfx1250): bf16 WMMA pipeline, f32 accumulate.
//   1) cvt x, Wq/Wk/Wv/Wo -> bf16
//   2) WMMA GEMM (async-LDS double-buffered W tiles): Q,K head-split; V transposed
//   3) flash attention: 64-key tiles, K/V^T staged to LDS via async DMA shared by
//      8 waves/block, double-buffered; online softmax; 32 WMMA per tile
//   4) WMMA GEMM output projection -> f32 d_out
// ---------------------------------------------------------------------------

typedef __bf16 bf16;
typedef __bf16 v16bf __attribute__((ext_vector_type(16)));
typedef __bf16 v8bf  __attribute__((ext_vector_type(8)));
typedef __bf16 v4bf  __attribute__((ext_vector_type(4)));
typedef float  v8f   __attribute__((ext_vector_type(8)));
typedef float  v4f   __attribute__((ext_vector_type(4)));

#define D_MODEL 2048
#define N_HEADS 16
#define D_HEAD  128
#define BATCH   4
#define SEQ     2048
#define ROWS    (BATCH*SEQ)   // 8192

// -------------------------------------------------- CDNA5 async LDS copy path
// The clang builtin's signature on this toolchain is a 4xi32 descriptor form
// (uncharacterized), so use inline asm per CDNA5 ISA:
//   GLOBAL_LOAD_ASYNC_TO_LDS_B128, GV mode: VDST = LDS byte addr VGPR,
//   VADDR = 64-bit global address. Tracked by ASYNCcnt.
// Generic pointers to LDS carry the LDS byte offset in their low 32 bits.
#define USE_ASYNC_ASM 1

static __device__ __forceinline__ void async_cp16(const bf16* g, bf16* l) {
#if USE_ASYNC_ASM
  asm volatile("global_load_async_to_lds_b128 %0, %1, off"
               :: "v"((unsigned)(size_t)l), "v"((unsigned long long)(size_t)g)
               : "memory");
#else
  *(v8bf*)l = *(const v8bf*)g;   // synchronous fallback (global load + ds store)
#endif
}

static __device__ __forceinline__ void async_wait() {
#if USE_ASYNC_ASM
  asm volatile("s_wait_asynccnt 0" ::: "memory");
#endif
}

static __device__ __forceinline__ v8f wmma_bf16(v16bf a, v16bf b, v8f c) {
  return __builtin_amdgcn_wmma_f32_16x16x32_bf16(false, a, false, b, (short)0, c,
                                                 false, false);
}

union AFrag { v16bf v; v8bf h[2]; };

// ---------------------------------------------------------------- conversion
__global__ void cvt_f32_bf16_k(const float* __restrict__ src,
                               bf16* __restrict__ dst, int n4) {
  int i = blockIdx.x * blockDim.x + threadIdx.x;
  int stride = gridDim.x * blockDim.x;
  for (; i < n4; i += stride) {
    v4f x = ((const v4f*)src)[i];
    v4bf y;
    y[0] = (bf16)x[0]; y[1] = (bf16)x[1]; y[2] = (bf16)x[2]; y[3] = (bf16)x[3];
    ((v4bf*)dst)[i] = y;
  }
}

// ------------------------------------------------------------------- GEMM
// out[m,n] = sum_k A[m,k] * W[n,k] + bias[n]   (A,W bf16 row-major, K contig)
// W tile (64 cols x 64 k) staged to LDS via async DMA, double-buffered, shared
// by the block's 8 waves (all share n0).
// MODE 0: f32 out (M,N);  MODE 1: bf16 (B,H,S,Dh);  MODE 2: bf16 (B,H,Dh,S)
template<int MODE>
__global__ __launch_bounds__(256)
void gemm_bf16_k(const bf16* __restrict__ A, const bf16* __restrict__ W,
                 const float* __restrict__ bias, void* __restrict__ outp) {
  __shared__ __align__(16) bf16 Wt[2][64][64];   // 8KB per buffer
  const int tid  = threadIdx.x;
  const int lane = tid & 31;
  const int wave = tid >> 5;
  const int half = lane >> 4;
  const int l16  = lane & 15;
  const int m0 = blockIdx.y * 128 + wave * 16;
  const int n0 = blockIdx.x * 64;

  v8f acc[4] = {};
  const bf16* Arow = A + (size_t)(m0 + l16) * D_MODEL;

  auto stage = [&](int buf, int k0) {
#pragma unroll
    for (int i = 0; i < 2; ++i) {
      const int c = tid * 2 + i;              // 512 x 16B chunks
      const int row = c >> 2, off = (c & 3) * 8;
      async_cp16(W + (size_t)(n0 + row) * D_MODEL + k0 + off, &Wt[buf][row][off]);
    }
  };

  stage(0, 0);
  async_wait();
  __syncthreads();

  int buf = 0;
  for (int kc = 0; kc < D_MODEL; kc += 64) {
    if (kc + 64 < D_MODEL) {
      stage(buf ^ 1, kc + 64);                 // overlap DMA with compute
      __builtin_prefetch(Arow + kc + 64, 0, 1);
    }
#pragma unroll
    for (int ks = 0; ks < 2; ++ks) {
      const int kk = kc + ks * 32;
      AFrag a;   // A layout: lane<16 -> K{0..15}; lane>=16 -> K{16..31}
      a.h[0] = *(const v8bf*)(Arow + kk + half * 16);
      a.h[1] = *(const v8bf*)(Arow + kk + half * 16 + 8);
#pragma unroll
      for (int j = 0; j < 4; ++j) {
        v16bf b = *(const v16bf*)&Wt[buf][j * 16 + l16][ks * 32 + half * 16];
        acc[j] = wmma_bf16(a.v, b, acc[j]);
      }
    }
    async_wait();
    __syncthreads();
    buf ^= 1;
  }

  const int mbase = m0 + half * 8;
#pragma unroll
  for (int j = 0; j < 4; ++j) {
    const int n = n0 + j * 16 + l16;
    const float bn = bias[n];
    if (MODE == 0) {
      float* out = (float*)outp;
#pragma unroll
      for (int r = 0; r < 8; ++r)
        out[(size_t)(mbase + r) * D_MODEL + n] = acc[j][r] + bn;
    } else if (MODE == 1) {
      bf16* out = (bf16*)outp;
      const int h = n >> 7, d = n & (D_HEAD - 1);
      const int b = mbase >> 11, s = mbase & (SEQ - 1);
#pragma unroll
      for (int r = 0; r < 8; ++r)
        out[((size_t)(b * N_HEADS + h) * SEQ + s + r) * D_HEAD + d] =
            (bf16)(acc[j][r] + bn);
    } else {
      bf16* out = (bf16*)outp;   // V^T (B,H,Dh,S): rows r contiguous in S
      const int h = n >> 7, d = n & (D_HEAD - 1);
      const int b = mbase >> 11, s = mbase & (SEQ - 1);
      v8bf pk;
#pragma unroll
      for (int r = 0; r < 8; ++r) pk[r] = (bf16)(acc[j][r] + bn);
      *(v8bf*)(out + ((size_t)(b * N_HEADS + h) * D_HEAD + d) * SEQ + s) = pk;
    }
  }
}

// --------------------------------------------------------------- attention
// One wave per 16-row Q block, 8 waves/block share (b,h). 64-key tiles of K
// and V^T staged to LDS with async DMA, double-buffered.
__global__ __launch_bounds__(256)
void flash_attn_k(const bf16* __restrict__ Q, const bf16* __restrict__ K,
                  const bf16* __restrict__ Vt, const int* __restrict__ mask,
                  bf16* __restrict__ ctx) {
  __shared__ __align__(16) bf16 Kt[2][64][128];    // 16KB per buffer
  __shared__ __align__(16) bf16 Vl[2][128][64];    // 16KB per buffer
  __shared__ __align__(16) bf16 ptile[8][16][64];  // 16KB, per-wave P tiles
  const int tid  = threadIdx.x;
  const int lane = tid & 31;
  const int wave = tid >> 5;
  const int half = lane >> 4;
  const int l16  = lane & 15;
  const int bh = blockIdx.y;                       // b*H + h
  const int q0 = blockIdx.x * 128 + wave * 16;

  v16bf qf[4];                                     // Q 16x128 as 4 A-frags
  {
    const bf16* Qrow = Q + ((size_t)bh * SEQ + q0 + l16) * D_HEAD;
#pragma unroll
    for (int j = 0; j < 4; ++j) {
      AFrag a;
      a.h[0] = *(const v8bf*)(Qrow + 32 * j + half * 16);
      a.h[1] = *(const v8bf*)(Qrow + 32 * j + half * 16 + 8);
      qf[j] = a.v;
    }
  }

  auto stage = [&](int buf, int kt) {
    const bf16* Kg = K + ((size_t)bh * SEQ + kt) * D_HEAD;   // 64x128
#pragma unroll
    for (int i = 0; i < 4; ++i) {
      const int c = tid * 4 + i;                 // 1024 x 16B chunks
      const int row = c >> 3, off = (c & 7) * 8;
      async_cp16(Kg + (size_t)row * D_HEAD + off, &Kt[buf][row][off]);
    }
    const bf16* Vg = Vt + (size_t)bh * D_HEAD * SEQ + kt;    // 128 rows x 64 k
#pragma unroll
    for (int i = 0; i < 4; ++i) {
      const int c = tid * 4 + i;
      const int row = c >> 3, off = (c & 7) * 8;
      async_cp16(Vg + (size_t)row * SEQ + off, &Vl[buf][row][off]);
    }
  };

  v8f acc[8] = {};
  float mst[8], lst[8];
#pragma unroll
  for (int r = 0; r < 8; ++r) { mst[r] = -1e30f; lst[r] = 0.f; }

  const float scale = 0.08838834764831845f;        // 1/sqrt(128)
  const int qrow = q0 + half * 8;

  stage(0, 0);
  async_wait();
  __syncthreads();

  int buf = 0;
  for (int kt = 0; kt < SEQ; kt += 64) {
    if (kt + 64 < SEQ) stage(buf ^ 1, kt + 64);   // overlap DMA with compute

    // ---- scores: Q(16x128) . K_tile(64x128)^T -> 4 C-frags (16 wmma)
    v8f sc[4] = {};
#pragma unroll
    for (int j = 0; j < 4; ++j) {
#pragma unroll
      for (int t = 0; t < 4; ++t) {
        v16bf b = *(const v16bf*)&Kt[buf][t * 16 + l16][32 * j + half * 16];
        sc[t] = wmma_bf16(qf[j], b, sc[t]);
      }
    }

    // ---- online softmax across 64 keys (row = qrow+r, col = l16 per frag)
#pragma unroll
    for (int r = 0; r < 8; ++r) {
      const int* mrow = mask + (size_t)(qrow + r) * SEQ + kt;
      float sv[4];
#pragma unroll
      for (int t = 0; t < 4; ++t)
        sv[t] = (mrow[t * 16 + l16] != 0) ? sc[t][r] * scale : -1e30f;
      float rowmax = fmaxf(fmaxf(sv[0], sv[1]), fmaxf(sv[2], sv[3]));
#pragma unroll
      for (int off = 1; off < 16; off <<= 1)
        rowmax = fmaxf(rowmax, __shfl_xor(rowmax, off, 16));
      const float mnew  = fmaxf(mst[r], rowmax);
      const float alpha = __expf(mst[r] - mnew);
      float p[4], rs = 0.f;
#pragma unroll
      for (int t = 0; t < 4; ++t) { p[t] = __expf(sv[t] - mnew); rs += p[t]; }
#pragma unroll
      for (int off = 1; off < 16; off <<= 1)
        rs += __shfl_xor(rs, off, 16);
      lst[r] = lst[r] * alpha + rs;
      mst[r] = mnew;
#pragma unroll
      for (int t = 0; t < 4; ++t)
        ptile[wave][half * 8 + r][t * 16 + l16] = (bf16)p[t];
#pragma unroll
      for (int dj = 0; dj < 8; ++dj) acc[dj][r] *= alpha;
    }

    // ---- P (16x64) as two A-frags via LDS relayout (same-wave DS: in-order)
    AFrag p0, p1;
    p0.h[0] = *(const v8bf*)&ptile[wave][l16][half * 16];
    p0.h[1] = *(const v8bf*)&ptile[wave][l16][half * 16 + 8];
    p1.h[0] = *(const v8bf*)&ptile[wave][l16][32 + half * 16];
    p1.h[1] = *(const v8bf*)&ptile[wave][l16][40 + half * 16];

    // ---- O += P . V_tile (16 wmma)
#pragma unroll
    for (int dj = 0; dj < 8; ++dj) {
      v16bf v0 = *(const v16bf*)&Vl[buf][16 * dj + l16][half * 16];
      v16bf v1 = *(const v16bf*)&Vl[buf][16 * dj + l16][32 + half * 16];
      acc[dj] = wmma_bf16(p0.v, v0, acc[dj]);
      acc[dj] = wmma_bf16(p1.v, v1, acc[dj]);
    }

    async_wait();       // next tile fully in LDS (this wave's DMA)
    __syncthreads();    // visible to all waves; all reads of buf done
    buf ^= 1;
  }

  const int b = bh >> 4, h = bh & (N_HEADS - 1);
#pragma unroll
  for (int dj = 0; dj < 8; ++dj) {
    const int d = h * D_HEAD + 16 * dj + l16;
#pragma unroll
    for (int r = 0; r < 8; ++r) {
      float o = acc[dj][r] / lst[r];
      ctx[(size_t)(b * SEQ + qrow + r) * D_MODEL + d] = (bf16)o;
    }
  }
}

// ----------------------------------------------------------------- launcher
extern "C" void kernel_launch(void* const* d_in, const int* in_sizes, int n_in,
                              void* d_out, int out_size, void* d_ws, size_t ws_size,
                              hipStream_t stream) {
  const float* x  = (const float*)d_in[0];
  const int* mask = (const int*)d_in[1];
  const float* Wq = (const float*)d_in[2];
  const float* bq = (const float*)d_in[3];
  const float* Wk = (const float*)d_in[4];
  const float* bk = (const float*)d_in[5];
  const float* Wv = (const float*)d_in[6];
  const float* bv = (const float*)d_in[7];
  const float* Wo = (const float*)d_in[8];
  const float* bo = (const float*)d_in[9];
  float* out = (float*)d_out;

  char* ws = (char*)d_ws;
  bf16* xb  = (bf16*)(ws);                    // 32 MiB  (B*S,D) bf16
  bf16* wqb = (bf16*)(ws + 33554432);         //  8 MiB each
  bf16* wkb = (bf16*)(ws + 41943040);
  bf16* wvb = (bf16*)(ws + 50331648);
  bf16* wob = (bf16*)(ws + 58720256);
  bf16* Qb  = (bf16*)(ws + 67108864);         // 32 MiB  (B,H,S,Dh)
  bf16* Kb  = (bf16*)(ws + 100663296);        // 32 MiB  (B,H,S,Dh)
  bf16* Vt  = (bf16*)(ws + 134217728);        // 32 MiB  (B,H,Dh,S)
  bf16* ctx = xb;  // x dead after projections; reuse as attention output

  cvt_f32_bf16_k<<<2048, 256, 0, stream>>>(x,  xb,  (ROWS * D_MODEL) / 4);
  cvt_f32_bf16_k<<<512,  256, 0, stream>>>(Wq, wqb, (D_MODEL * D_MODEL) / 4);
  cvt_f32_bf16_k<<<512,  256, 0, stream>>>(Wk, wkb, (D_MODEL * D_MODEL) / 4);
  cvt_f32_bf16_k<<<512,  256, 0, stream>>>(Wv, wvb, (D_MODEL * D_MODEL) / 4);
  cvt_f32_bf16_k<<<512,  256, 0, stream>>>(Wo, wob, (D_MODEL * D_MODEL) / 4);

  dim3 ggrid(D_MODEL / 64, ROWS / 128);
  gemm_bf16_k<1><<<ggrid, 256, 0, stream>>>(xb, wqb, bq, Qb);
  gemm_bf16_k<1><<<ggrid, 256, 0, stream>>>(xb, wkb, bk, Kb);
  gemm_bf16_k<2><<<ggrid, 256, 0, stream>>>(xb, wvb, bv, Vt);

  dim3 fgrid(SEQ / 128, BATCH * N_HEADS);
  flash_attn_k<<<fgrid, 256, 0, stream>>>(Qb, Kb, Vt, mask, ctx);

  gemm_bf16_k<0><<<ggrid, 256, 0, stream>>>(ctx, wob, bo, out);
}
